// PointDecoder2_14070312861807
// MI455X (gfx1250) — compile-verified, matrix operand order checked by reading, not run on hardware
//
#include <hip/hip_runtime.h>
#include <hip/hip_bf16.h>
#include <math.h>

typedef __attribute__((ext_vector_type(16))) _Float16 v16h;
typedef __attribute__((ext_vector_type(8)))  float    v8f;
typedef __attribute__((ext_vector_type(4)))  int      v4i;

#define BSZ   8
#define NPT   1024
#define DIM   256
#define NH    8
#define DHD   32
#define FFD   1024
#define NLAY  4
#define NTOK  (BSZ * NPT)   // 8192

// ---------------- CDNA5 async global->LDS copy (guarded) -----------------

#if defined(__has_builtin)
# if __has_builtin(__builtin_amdgcn_global_load_async_to_lds_b128)
#  define HAVE_ASYNC_COPY 1
# endif
# if __has_builtin(__builtin_amdgcn_s_wait_asynccnt)
#  define HAVE_WAIT_ASYNC 1
# endif
#endif
#ifndef HAVE_ASYNC_COPY
# define HAVE_ASYNC_COPY 0
#endif
#ifndef HAVE_WAIT_ASYNC
# define HAVE_WAIT_ASYNC 0
#endif

typedef __attribute__((address_space(1))) v4i* gptr_v4i;
typedef __attribute__((address_space(3))) v4i* lptr_v4i;

// copy 16 bytes (8 f16) global -> LDS, ASYNCcnt-tracked when available
__device__ inline void copy16_g2l(const _Float16* g, _Float16* l) {
#if HAVE_ASYNC_COPY
  __builtin_amdgcn_global_load_async_to_lds_b128((gptr_v4i)(void*)g,
                                                 (lptr_v4i)(void*)l, 0, 0);
#else
#pragma unroll
  for (int e = 0; e < 8; ++e) l[e] = g[e];
#endif
}

__device__ inline void async_wait0() {
#if HAVE_ASYNC_COPY
# if HAVE_WAIT_ASYNC
  __builtin_amdgcn_s_wait_asynccnt(0);
# else
  asm volatile("s_wait_asynccnt 0" ::: "memory");
# endif
#endif
}

// ---------------- WMMA fragment helpers (ISA 7.12.2 layouts) -------------

// A-frag / transposed-B-frag: element (k, r) read from S[r*ld + k].
// Lane: r = r0 + (lane&15); contiguous pairs -> ds_load_b128 x2.
__device__ inline v16h frag_row16(const _Float16* S, int r0, int ld, int lane) {
  const _Float16* p = S + (size_t)(r0 + (lane & 15)) * ld + ((lane >> 4) << 3);
  v16h a;
#pragma unroll
  for (int e = 0; e < 8; ++e) { a[e] = p[e]; a[8 + e] = p[16 + e]; }
  return a;
}

// B-frag from k-major storage: element (k, c) read from S[k*ld + c] (strided).
__device__ inline v16h frag_col16(const _Float16* S, int col0, int ld, int lane) {
  const _Float16* p = S + col0 + (lane & 15);
  int kh = (lane >> 4) << 3;
  v16h b;
#pragma unroll
  for (int i = 0; i < 8; ++i) {
    int kk = ((i & 4) << 2) + kh + ((i & 3) << 1);
    b[2 * i]     = p[(size_t)kk * ld];
    b[2 * i + 1] = p[(size_t)(kk + 1) * ld];
  }
  return b;
}

__device__ inline v8f wmma16(v16h a, v16h b, v8f c) {
  return __builtin_amdgcn_wmma_f32_16x16x32_f16(false, a, false, b, (short)0, c,
                                                false, false);
}

__device__ inline v8f vzero8() {
  v8f z;
#pragma unroll
  for (int i = 0; i < 8; ++i) z[i] = 0.0f;
  return z;
}

// ---------------- Weight convert + transpose: W[K][N] f32 -> Wt[N][K] f16 -

__global__ __launch_bounds__(256)
void cvt_w_t(const float* __restrict__ W, _Float16* __restrict__ Wt,
             int K, int N) {
  int id = blockIdx.x * 256 + threadIdx.x;
  if (id >= K * N) return;
  int k = id / N, n = id - k * N;
  Wt[(size_t)n * K + k] = (_Float16)W[id];
}

// ---------------- Tiled WMMA GEMM, all-f16 operands ----------------------
// A: MxK f16 row-major. Bt: NxK f16 (N-major / transposed).
// Out: Cf (f32, +bias +res) or Ch (f16, +bias, relu).

#define TM 64
#define TN 128
#define TK 32

__global__ __launch_bounds__(256)
void gemm_wmma(const _Float16* __restrict__ A, const _Float16* __restrict__ Bt,
               const float* __restrict__ bias, const float* __restrict__ res,
               float* __restrict__ Cf, _Float16* __restrict__ Ch,
               int M, int Nn, int K, int relu) {
  __shared__ __align__(16) _Float16 As[TM][TK];   // 64x32, 64B rows
  __shared__ __align__(16) _Float16 Bs[TN][TK];   // 128x32, N-major

  int tid  = threadIdx.x;
  int lane = tid & 31;
  int wave = tid >> 5;
  int wrow = wave >> 2;          // 0..1
  int wcol = wave & 3;           // 0..3
  int row0 = blockIdx.y * TM;
  int col0 = blockIdx.x * TN;

  v8f acc[2][2];
#pragma unroll
  for (int i = 0; i < 2; ++i)
#pragma unroll
    for (int j = 0; j < 2; ++j) acc[i][j] = vzero8();

  int ar = tid >> 2;             // 0..63
  int ac = (tid & 3) * 8;        // chunk of 8 f16 (16B)
  int brn = tid >> 1;            // 0..127 (N row of Bt tile)
  int bcs = (tid & 1) * 16;      // 2x16B per thread

  for (int k0 = 0; k0 < K; k0 += TK) {
    copy16_g2l(A + (size_t)(row0 + ar) * K + k0 + ac, &As[ar][ac]);
    const _Float16* bg = Bt + (size_t)(col0 + brn) * K + k0 + bcs;
    copy16_g2l(bg,     &Bs[brn][bcs]);
    copy16_g2l(bg + 8, &Bs[brn][bcs + 8]);
    if (k0 + TK < K) {
      __builtin_prefetch(A + (size_t)(row0 + ar) * K + k0 + TK + ac, 0, 0);
      __builtin_prefetch(bg + TK, 0, 0);
    }
    async_wait0();
    __syncthreads();

    v16h af[2], bf[2];
#pragma unroll
    for (int i = 0; i < 2; ++i)
      af[i] = frag_row16(&As[0][0], wrow * 32 + 16 * i, TK, lane);
#pragma unroll
    for (int j = 0; j < 2; ++j)
      bf[j] = frag_row16(&Bs[0][0], wcol * 32 + 16 * j, TK, lane);
#pragma unroll
    for (int i = 0; i < 2; ++i)
#pragma unroll
      for (int j = 0; j < 2; ++j) acc[i][j] = wmma16(af[i], bf[j], acc[i][j]);
    __syncthreads();
  }

  int cl = lane & 15;
  int rh = (lane >> 4) * 8;
#pragma unroll
  for (int i = 0; i < 2; ++i) {
#pragma unroll
    for (int j = 0; j < 2; ++j) {
      int cc = col0 + wcol * 32 + j * 16 + cl;
      float bval = bias ? bias[cc] : 0.0f;
#pragma unroll
      for (int v = 0; v < 8; ++v) {
        int r = row0 + wrow * 32 + i * 16 + rh + v;
        float val = acc[i][j][v] + bval;
        if (relu) val = fmaxf(val, 0.0f);
        if (Cf) {
          if (res) val += res[(size_t)r * Nn + cc];
          Cf[(size_t)r * Nn + cc] = val;
        } else {
          Ch[(size_t)r * Nn + cc] = (_Float16)val;
        }
      }
    }
  }
}

// ---------------- LayerNorm: f32 in, f16 out, one token per wave ---------

__global__ __launch_bounds__(256)
void ln_kernel(const float* __restrict__ x, const float* __restrict__ s,
               const float* __restrict__ b, _Float16* __restrict__ y) {
  int wave = threadIdx.x >> 5;
  int lane = threadIdx.x & 31;
  int t = blockIdx.x * 8 + wave;
  const float* xr = x + (size_t)t * DIM;
  float v[8];
  float sum = 0.0f;
#pragma unroll
  for (int e = 0; e < 8; ++e) { v[e] = xr[e * 32 + lane]; sum += v[e]; }
  for (int m = 16; m > 0; m >>= 1) sum += __shfl_xor(sum, m, 32);
  float mean = sum * (1.0f / DIM);
  float sq = 0.0f;
#pragma unroll
  for (int e = 0; e < 8; ++e) { float d = v[e] - mean; sq += d * d; }
  for (int m = 16; m > 0; m >>= 1) sq += __shfl_xor(sq, m, 32);
  float inv = rsqrtf(sq * (1.0f / DIM) + 1e-5f);
  _Float16* yr = y + (size_t)t * DIM;
#pragma unroll
  for (int e = 0; e < 8; ++e) {
    int c = e * 32 + lane;
    yr[c] = (_Float16)((v[e] - mean) * inv * s[c] + b[c]);
  }
}

// ---------------- Flash attention, f16 qkv, WMMA (DH=32 == K) ------------
// grid.x = BSZ*NH, grid.y = NPT/64, block = 128 (4 waves, 16 queries/wave)

__global__ __launch_bounds__(128)
void attn_kernel(const _Float16* __restrict__ qkv, _Float16* __restrict__ out) {
  __shared__ __align__(16) _Float16 Kl[32][32];       // [key][dh] (= B^T of QK^T)
  __shared__ __align__(16) _Float16 Vl[32][32];       // [key][dh]
  __shared__ __align__(16) _Float16 Pl[4][16][32];    // per-wave P (16q x 32k)

  int tid  = threadIdx.x;
  int lane = tid & 31;
  int wave = tid >> 5;
  int b  = blockIdx.x >> 3;
  int hh = blockIdx.x & 7;
  int q0 = blockIdx.y * 64 + wave * 16;
  const float scale = 0.17677669529663689f;   // 1/sqrt(32)

  // Q fragment straight from global (contiguous-pair layout, ld = 768)
  v16h qf = frag_row16(qkv + (size_t)(b * NPT + q0) * (3 * DIM) + hh * DHD,
                       0, 3 * DIM, lane);

  float mrow[8], lrow[8];
#pragma unroll
  for (int v = 0; v < 8; ++v) { mrow[v] = -1e30f; lrow[v] = 0.0f; }
  v8f oacc[2];
  oacc[0] = vzero8(); oacc[1] = vzero8();

  int cl = lane & 15;
  int rh = (lane >> 4) * 8;

  for (int j0 = 0; j0 < NPT; j0 += 32) {
    // stage K and V tiles (32 keys x 32 dh, f16) via async LDS copies
    int key = tid >> 2;
    int ch  = (tid & 3) * 8;
    size_t tok = (size_t)(b * NPT + j0 + key) * (3 * DIM) + hh * DHD + ch;
    copy16_g2l(qkv + tok + DIM,     &Kl[key][ch]);
    copy16_g2l(qkv + tok + 2 * DIM, &Vl[key][ch]);
    async_wait0();
    __syncthreads();

    // scores for 32 keys: K stored [key][dh] == transposed-B layout
    v8f sacc[2];
    sacc[0] = vzero8(); sacc[1] = vzero8();
#pragma unroll
    for (int t = 0; t < 2; ++t) {
      v16h kf = frag_row16(&Kl[0][0], t * 16, 32, lane);
      sacc[t] = wmma16(qf, kf, sacc[t]);
    }

    // online softmax (row reductions across 16-lane half-groups)
#pragma unroll
    for (int v = 0; v < 8; ++v) {
      float s0 = sacc[0][v] * scale;
      float s1 = sacc[1][v] * scale;
      float mx = fmaxf(s0, s1);
      for (int m = 8; m > 0; m >>= 1) mx = fmaxf(mx, __shfl_xor(mx, m, 32));
      float mn = fmaxf(mrow[v], mx);
      float c  = __expf(mrow[v] - mn);
      float p0 = __expf(s0 - mn);
      float p1 = __expf(s1 - mn);
      float rs = p0 + p1;
      for (int m = 8; m > 0; m >>= 1) rs += __shfl_xor(rs, m, 32);
      lrow[v] = lrow[v] * c + rs;
      mrow[v] = mn;
      sacc[0][v] = p0; sacc[1][v] = p1;
      oacc[0][v] *= c; oacc[1][v] *= c;
    }

    // redistribute P (C-frag layout) into A-frag layout via LDS
#pragma unroll
    for (int t = 0; t < 2; ++t)
#pragma unroll
      for (int v = 0; v < 8; ++v)
        Pl[wave][rh + v][t * 16 + cl] = (_Float16)sacc[t][v];
    __syncthreads();

    v16h pf = frag_row16(&Pl[wave][0][0], 0, 32, lane);
#pragma unroll
    for (int t = 0; t < 2; ++t) {
      v16h vf = frag_col16(&Vl[0][0], t * 16, 32, lane);
      oacc[t] = wmma16(pf, vf, oacc[t]);
    }
    __syncthreads();
  }

  // normalize + write O (f16) in [b, n, h*dh] layout
#pragma unroll
  for (int t = 0; t < 2; ++t)
#pragma unroll
    for (int v = 0; v < 8; ++v) {
      int q = q0 + rh + v;
      out[(size_t)(b * NPT + q) * DIM + hh * DHD + t * 16 + cl] =
          (_Float16)(oacc[t][v] / lrow[v]);
    }
}

// ---------------- Preprocess: label fill + ball points (hash RNG) --------

__global__ __launch_bounds__(256)
void prep_kernel(const float* __restrict__ input_point,
                 const int* __restrict__ input_label,
                 float* __restrict__ pts, int* __restrict__ label) {
  int t = blockIdx.x * blockDim.x + threadIdx.x;
  if (t >= NTOK) return;
  int b = t >> 10;
  int n = t & 1023;
  int lab = input_label[t];
  int pad = (lab != -2);
  int outlab = lab;
  if (!pad) {
    unsigned mask = 0;
    for (int j = 0; j < NPT; ++j) {
      int lj = input_label[b * NPT + j];
      if (lj >= 0) mask |= 1u << lj;
    }
    unsigned s = (unsigned)t * 2654435761u ^ 0x9e3779b9u;
    s ^= s << 13; s ^= s >> 17; s ^= s << 5;
    int cnt = __popc(mask & 0xFu);
    int pick = cnt ? (int)(s % (unsigned)cnt) : 0;
    int p = 0;
    for (int q = 0; q < 4; ++q) {
      if ((mask >> q) & 1u) { if (pick == 0) { p = q; break; } --pick; }
    }
    outlab = p;
  }
  label[t] = outlab;

  unsigned s2 = (unsigned)n * 1664525u + 1013904223u;
  float u[5];
#pragma unroll
  for (int i = 0; i < 5; ++i) {
    s2 = s2 * 1664525u + 1013904223u;
    u[i] = (float)(s2 >> 8) * (1.0f / 16777216.0f);
  }
  float r1 = sqrtf(-2.0f * __logf(u[0] + 1e-7f));
  float g0 = r1 * __cosf(6.2831853f * u[1]);
  float g1 = r1 * __sinf(6.2831853f * u[1]);
  float r2 = sqrtf(-2.0f * __logf(u[2] + 1e-7f));
  float g2 = r2 * __cosf(6.2831853f * u[3]);
  float nrm = rsqrtf(g0 * g0 + g1 * g1 + g2 * g2 + 1e-12f);
  float rad = powf(u[4], 1.0f / 3.0f) * 0.5f;
  float bx = g0 * nrm * rad, by = g1 * nrm * rad, bz = g2 * nrm * rad;
  float pf = pad ? 1.0f : 0.0f;
  pts[t * 3 + 0] = input_point[t * 3 + 0] * pf + bx * (1.0f - pf);
  pts[t * 3 + 1] = input_point[t * 3 + 1] * pf + by * (1.0f - pf);
  pts[t * 3 + 2] = input_point[t * 3 + 2] * pf + bz * (1.0f - pf);
}

// ---------------- Embedding ----------------------------------------------

__global__ __launch_bounds__(256)
void embed_kernel(const float* __restrict__ pts, const int* __restrict__ label,
                  const int* __restrict__ input_label,
                  const float* __restrict__ enc, const float* __restrict__ eW,
                  const float* __restrict__ eb, const float* __restrict__ lemb,
                  float* __restrict__ x) {
  int t = blockIdx.x;
  int c = threadIdx.x;
  int b = t >> 10;
  int lab = label[t];
  int pad = (input_label[t] != -2);
  float acc = eb[c];
#pragma unroll
  for (int j = 0; j < 3; ++j) acc += pts[t * 3 + j] * eW[j * DIM + c];
  const float* pf = enc + (size_t)(b * 4 + lab) * DIM;
  for (int j = 0; j < DIM; ++j) acc += pf[j] * eW[(3 + j) * DIM + c];
  acc += (pad ? 0.0f : 1.0f) * eW[259 * DIM + c];
  acc += (pad ? 1.0f : 0.0f) * eW[260 * DIM + c];
  acc += lemb[lab * DIM + c];
  x[(size_t)t * DIM + c] = acc;
}

// ---------------- Head MLP: ln'd x(f16) -> 128 -> 32 -> 3 ----------------

__global__ __launch_bounds__(256)
void head_kernel(const _Float16* __restrict__ xln, const float* __restrict__ pts,
                 const int* __restrict__ label, const float* __restrict__ h1W,
                 const float* __restrict__ h1b, const float* __restrict__ h2W,
                 const float* __restrict__ h2b, const float* __restrict__ h3W,
                 const float* __restrict__ h3b, float* __restrict__ dout) {
  __shared__ float xs[8][DIM];
  __shared__ float y1s[8][128];
  __shared__ float y2s[8][32];
  int wave = threadIdx.x >> 5;
  int lane = threadIdx.x & 31;
  int t = blockIdx.x * 8 + wave;

#pragma unroll
  for (int e = 0; e < 8; ++e)
    xs[wave][e * 32 + lane] = (float)xln[(size_t)t * DIM + e * 32 + lane];
  __syncthreads();

  float v1[4];
#pragma unroll
  for (int cc = 0; cc < 4; ++cc) {
    int c = lane * 4 + cc;
    float a = h1b[c];
    for (int j = 0; j < DIM; ++j) a += xs[wave][j] * h1W[j * 128 + c];
    v1[cc] = a;
  }
  float sum = v1[0] + v1[1] + v1[2] + v1[3];
  for (int m = 16; m > 0; m >>= 1) sum += __shfl_xor(sum, m, 32);
  float mean = sum * (1.0f / 128.0f);
  float sq = 0.0f;
#pragma unroll
  for (int cc = 0; cc < 4; ++cc) { float d = v1[cc] - mean; sq += d * d; }
  for (int m = 16; m > 0; m >>= 1) sq += __shfl_xor(sq, m, 32);
  float inv = rsqrtf(sq * (1.0f / 128.0f) + 1e-5f);
#pragma unroll
  for (int cc = 0; cc < 4; ++cc)
    y1s[wave][lane * 4 + cc] = fmaxf((v1[cc] - mean) * inv, 0.0f);
  __syncthreads();

  float a2 = h2b[lane];
  for (int j = 0; j < 128; ++j) a2 += y1s[wave][j] * h2W[j * 32 + lane];
  float s2 = a2;
  for (int m = 16; m > 0; m >>= 1) s2 += __shfl_xor(s2, m, 32);
  float m2 = s2 * (1.0f / 32.0f);
  float d2 = a2 - m2;
  float q2 = d2 * d2;
  for (int m = 16; m > 0; m >>= 1) q2 += __shfl_xor(q2, m, 32);
  y2s[wave][lane] = fmaxf(d2 * rsqrtf(q2 * (1.0f / 32.0f) + 1e-5f), 0.0f);
  __syncthreads();

  int b = t >> 10;
  int n = t & 1023;
  if (n >= 4) {
    if (lane < 3) {
      float a3 = h3b[lane];
      for (int j = 0; j < 32; ++j) a3 += y2s[wave][j] * h3W[j * 3 + lane];
      dout[((size_t)b * (NPT - 4) + (n - 4)) * 3 + lane] =
          pts[t * 3 + lane] + a3;
    }
    if (lane == 3) {
      dout[(size_t)BSZ * (NPT - 4) * 3 + (size_t)b * (NPT - 4) + (n - 4)] =
          (float)label[t];
    }
  }
}

// ---------------- Host orchestration -------------------------------------

extern "C" void kernel_launch(void* const* d_in, const int* in_sizes, int n_in,
                              void* d_out, int out_size, void* d_ws, size_t ws_size,
                              hipStream_t stream) {
  const float* enc    = (const float*)d_in[0];
  const float* ipt    = (const float*)d_in[1];
  const int*   ilab   = (const int*)d_in[2];
  const float* eW     = (const float*)d_in[3];
  const float* eb     = (const float*)d_in[4];
  const float* lemb   = (const float*)d_in[5];
  const float* Wqkv   = (const float*)d_in[6];
  const float* bqkv   = (const float*)d_in[7];
  const float* Wo     = (const float*)d_in[8];
  const float* bo     = (const float*)d_in[9];
  const float* ln1s   = (const float*)d_in[10];
  const float* ln1b   = (const float*)d_in[11];
  const float* ln2s   = (const float*)d_in[12];
  const float* ln2b   = (const float*)d_in[13];
  const float* ffW1   = (const float*)d_in[14];
  const float* ffb1   = (const float*)d_in[15];
  const float* ffW2   = (const float*)d_in[16];
  const float* ffb2   = (const float*)d_in[17];
  const float* lnfs   = (const float*)d_in[18];
  const float* lnfb   = (const float*)d_in[19];
  const float* h1W    = (const float*)d_in[20];
  const float* h1b    = (const float*)d_in[21];
  const float* h2W    = (const float*)d_in[22];
  const float* h2b    = (const float*)d_in[23];
  const float* h3W    = (const float*)d_in[24];
  const float* h3b    = (const float*)d_in[25];

  char* base = (char*)d_ws;
  size_t off = 0;
  auto alloc = [&](size_t bytes) {
    void* p = base + off;
    off += (bytes + 255) & ~(size_t)255;
    return p;
  };

  float*     pts  = (float*)alloc((size_t)NTOK * 3 * 4);
  int*       lab  = (int*)alloc((size_t)NTOK * 4);
  float*     x    = (float*)alloc((size_t)NTOK * DIM * 4);
  _Float16*  h    = (_Float16*)alloc((size_t)NTOK * DIM * 2);
  _Float16*  qkv  = (_Float16*)alloc((size_t)NTOK * 3 * DIM * 2);
  _Float16*  att  = (_Float16*)alloc((size_t)NTOK * DIM * 2);
  _Float16*  ff1  = (_Float16*)alloc((size_t)NTOK * FFD * 2);
  _Float16*  WqkvT = (_Float16*)alloc((size_t)NLAY * DIM * 3 * DIM * 2);
  _Float16*  WoT   = (_Float16*)alloc((size_t)NLAY * DIM * DIM * 2);
  _Float16*  W1T   = (_Float16*)alloc((size_t)NLAY * DIM * FFD * 2);
  _Float16*  W2T   = (_Float16*)alloc((size_t)NLAY * FFD * DIM * 2);

  // one-time (per call) weight convert+transpose to f16 N-major
  for (int i = 0; i < NLAY; ++i) {
    cvt_w_t<<<(DIM * 3 * DIM + 255) / 256, 256, 0, stream>>>(
        Wqkv + (size_t)i * DIM * 3 * DIM, WqkvT + (size_t)i * DIM * 3 * DIM,
        DIM, 3 * DIM);
    cvt_w_t<<<(DIM * DIM + 255) / 256, 256, 0, stream>>>(
        Wo + (size_t)i * DIM * DIM, WoT + (size_t)i * DIM * DIM, DIM, DIM);
    cvt_w_t<<<(DIM * FFD + 255) / 256, 256, 0, stream>>>(
        ffW1 + (size_t)i * DIM * FFD, W1T + (size_t)i * DIM * FFD, DIM, FFD);
    cvt_w_t<<<(FFD * DIM + 255) / 256, 256, 0, stream>>>(
        ffW2 + (size_t)i * FFD * DIM, W2T + (size_t)i * FFD * DIM, FFD, DIM);
  }

  prep_kernel<<<NTOK / 256, 256, 0, stream>>>(ipt, ilab, pts, lab);
  embed_kernel<<<NTOK, DIM, 0, stream>>>(pts, lab, ilab, enc, eW, eb, lemb, x);

  for (int i = 0; i < NLAY; ++i) {
    ln_kernel<<<NTOK / 8, 256, 0, stream>>>(x, ln1s + i * DIM, ln1b + i * DIM, h);
    gemm_wmma<<<dim3((3 * DIM) / TN, NTOK / TM), 256, 0, stream>>>(
        h, WqkvT + (size_t)i * DIM * 3 * DIM, bqkv + (size_t)i * 3 * DIM,
        nullptr, nullptr, qkv, NTOK, 3 * DIM, DIM, 0);
    attn_kernel<<<dim3(BSZ * NH, NPT / 64), 128, 0, stream>>>(qkv, att);
    gemm_wmma<<<dim3(DIM / TN, NTOK / TM), 256, 0, stream>>>(
        att, WoT + (size_t)i * DIM * DIM, bo + (size_t)i * DIM, x, x,
        nullptr, NTOK, DIM, DIM, 0);
    ln_kernel<<<NTOK / 8, 256, 0, stream>>>(x, ln2s + i * DIM, ln2b + i * DIM, h);
    gemm_wmma<<<dim3(FFD / TN, NTOK / TM), 256, 0, stream>>>(
        h, W1T + (size_t)i * DIM * FFD, ffb1 + (size_t)i * FFD,
        nullptr, nullptr, ff1, NTOK, FFD, DIM, 1);
    gemm_wmma<<<dim3(DIM / TN, NTOK / TM), 256, 0, stream>>>(
        ff1, W2T + (size_t)i * FFD * DIM, ffb2 + (size_t)i * DIM, x, x,
        nullptr, NTOK, DIM, FFD, 0);
  }

  ln_kernel<<<NTOK / 8, 256, 0, stream>>>(x, lnfs, lnfb, h);
  head_kernel<<<NTOK / 8, 256, 0, stream>>>(h, pts, lab, h1W, h1b, h2W, h2b,
                                            h3W, h3b, (float*)d_out);
}
